// CharLSTM_13305808683665
// MI455X (gfx1250) — matrix operand order repaired and checked by reading, not compiled, not run on hardware
//
#include <hip/hip_runtime.h>
#include <math.h>

// ---------------------------------------------------------------------------
// CharLSTM on MI455X (gfx1250): persistent-grid sequential LSTM, v2.
//   - fused gate weights row-interleaved (f,i,g,o per unit) so each 16-row
//     WMMA tile yields 4 complete units -> elementwise update stays in the
//     owning wave's registers (c,h live in VGPRs for all 16384 steps)
//   - z vector ping-pong (read t&1, write (t+1)&1) -> ONE grid barrier/step
//   - logits WG runs one step delayed, writes out[] directly with fused bias
//   - weights bf16 (42 MB, L2-resident), matvec via v_wmma_f32_16x16x32_bf16
// ---------------------------------------------------------------------------

#define T_SEQ 16384
#define EMB   512
#define HID   2048
#define NCH   128
#define KZ    (EMB + HID)     // 2560
#define GROWS (4 * HID)       // 8192 fused gate rows
#define GTILES (GROWS / 16)   // 512
#define LTILES (NCH / 16)     // 8
#define NWG   65              // 64 gate WGs + 1 logits WG
#define BLK   256

typedef __attribute__((ext_vector_type(16))) __bf16 v16bf;
typedef __attribute__((ext_vector_type(8)))  __bf16 v8bf;
typedef __attribute__((ext_vector_type(8)))  float  v8f;

__device__ __forceinline__ float sigmoidf_(float x) {
    return 1.0f / (1.0f + __expf(-x));
}

// ---------------------------------------------------------------------------
// Grid-wide barrier: bar[0] = arrival counter, bar[1] = epoch.
// ---------------------------------------------------------------------------
__device__ __forceinline__ void grid_barrier(unsigned* bar, unsigned nwg) {
    __syncthreads();
    if (threadIdx.x == 0) {
        __threadfence();
        unsigned* cnt = bar;
        unsigned* ep  = bar + 1;
        unsigned e = __atomic_load_n(ep, __ATOMIC_ACQUIRE);
        unsigned arrived = __atomic_fetch_add(cnt, 1u, __ATOMIC_ACQ_REL);
        if (arrived == nwg - 1u) {
            __atomic_store_n(cnt, 0u, __ATOMIC_RELAXED);
            __atomic_fetch_add(ep, 1u, __ATOMIC_ACQ_REL);
        } else {
            while (__atomic_load_n(ep, __ATOMIC_ACQUIRE) == e) {
                __builtin_amdgcn_s_sleep(2);
            }
        }
    }
    __syncthreads();
}

// ---------------------------------------------------------------------------
// 16-row matvec tile via WMMA bf16:  acc[16] = W[16 x K] * z[K]
// A fragment (16-bit 16x32, ISA 7.12.2): lane l -> row l&15,
//   K = {kb..kb+7} in v0-3 and {kb+16..kb+23} in v4-7, kb = (l>=16)*8.
// B fragment: same K order as A (kb..kb+7, kb+16..kb+23); the same z
//   chunk is fed to every column, so all 16 D columns equal the matvec.
// D layout: VGPR j = row j (lanes 0-15) / row j+8 (lanes 16-31).
// ---------------------------------------------------------------------------
__device__ __forceinline__ v8f wmma_matvec_tile(const __bf16* __restrict__ Wtile,
                                                int ldw,
                                                const __bf16* __restrict__ z,
                                                int K, int lane) {
    const int r    = lane & 15;
    const int half = lane >> 4;
    const __bf16* wp = Wtile + (size_t)r * ldw + half * 8;
    const __bf16* zp = z + half * 8;
    v8f acc = {};
    for (int k0 = 0; k0 < K; k0 += 32) {
        v8bf alo = *(const v8bf*)(wp + k0);
        v8bf ahi = *(const v8bf*)(wp + k0 + 16);
        v8bf blo = *(const v8bf*)(zp + k0);
        v8bf bhi = *(const v8bf*)(zp + k0 + 16);
        v16bf a, b;
#pragma unroll
        for (int i = 0; i < 8; ++i) {
            a[i] = alo[i]; a[i + 8] = ahi[i];
            b[i] = blo[i]; b[i + 8] = bhi[i];
        }
        acc = __builtin_amdgcn_wmma_f32_16x16x32_bf16(
                  false, a, false, b, (short)0, acc, false, false);
    }
    return acc;
}

// ---------------------------------------------------------------------------
// Setup: fused+interleaved bf16 gate weights/biases, Wfc bf16, pre-embedded
// inputs, z0 = [x_0 | 0], barrier words (ws is poisoned before timing).
// Fused row r = gate (r%4 : 0=f,1=i,2=g,3=o) of unit (r/4).
// ---------------------------------------------------------------------------
__global__ void k_prep(const int* __restrict__ seq, const float* __restrict__ emb,
                       const float* __restrict__ Wf, const float* __restrict__ Wi,
                       const float* __restrict__ Wg, const float* __restrict__ Wo,
                       const float* __restrict__ Wfc,
                       const float* __restrict__ bf_, const float* __restrict__ bi_,
                       const float* __restrict__ bg_, const float* __restrict__ bo_,
                       __bf16* __restrict__ Wz, __bf16* __restrict__ Wfcb,
                       __bf16* __restrict__ xemb, __bf16* __restrict__ z0,
                       float* __restrict__ bz, unsigned* __restrict__ bar) {
    const size_t stride = (size_t)gridDim.x * blockDim.x;
    const size_t tid0   = (size_t)blockIdx.x * blockDim.x + threadIdx.x;

    const size_t NW = (size_t)GROWS * KZ;
    for (size_t i = tid0; i < NW; i += stride) {
        size_t row = i / KZ, col = i - row * KZ;
        size_t u = row >> 2, g = row & 3;
        const float* src = (g == 0) ? Wf : (g == 1) ? Wi : (g == 2) ? Wg : Wo;
        Wz[i] = (__bf16)src[u * KZ + col];
    }
    for (size_t i = tid0; i < (size_t)GROWS; i += stride) {
        size_t u = i >> 2, g = i & 3;
        const float* src = (g == 0) ? bf_ : (g == 1) ? bi_ : (g == 2) ? bg_ : bo_;
        bz[i] = src[u];
    }
    const size_t NF = (size_t)NCH * HID;
    for (size_t i = tid0; i < NF; i += stride) Wfcb[i] = (__bf16)Wfc[i];
    const size_t NE = (size_t)T_SEQ * EMB;
    for (size_t i = tid0; i < NE; i += stride) {
        size_t t = i / EMB, e = i - t * EMB;
        xemb[i] = (__bf16)emb[(size_t)seq[t] * EMB + e];
    }
    for (size_t i = tid0; i < KZ; i += stride)
        z0[i] = (i < EMB) ? (__bf16)emb[(size_t)seq[0] * EMB + i] : (__bf16)0.0f;
    if (tid0 == 0) { bar[0] = 0u; bar[1] = 0u; }
}

// ---------------------------------------------------------------------------
// Persistent LSTM: 65 WGs x 256 threads = 520 waves, ONE barrier per step.
// Blocks 0..63: gate waves (tile = global wave id). Block 64: logits waves.
// ---------------------------------------------------------------------------
__global__ void __launch_bounds__(BLK)
k_lstm(const __bf16* __restrict__ Wz, const __bf16* __restrict__ Wfcb,
       const __bf16* __restrict__ xemb,
       __bf16* __restrict__ z0, __bf16* __restrict__ z1,
       const float* __restrict__ bz, const float* __restrict__ bfc_,
       unsigned* __restrict__ bar, float* __restrict__ out) {
    const int gtid = blockIdx.x * BLK + threadIdx.x;
    const int lane = threadIdx.x & 31;
    const int wv   = gtid >> 5;           // 0..519
    const int half = lane >> 4;
    const bool extract = (lane & 15) == 0;   // lanes 0 and 16 hold D rows

    if (wv < GTILES) {
        // -------- gate wave: 16 fused rows = units 4wv..4wv+3 --------
        const float* bp = bz + wv * 16 + half * 8;
        float bb[8];
#pragma unroll
        for (int j = 0; j < 8; ++j) bb[j] = bp[j];
        const __bf16* Wtile = Wz + (size_t)wv * 16 * KZ;
        const int ua = 4 * wv + half * 2;    // first of my 2 units (lane 0/16)
        float c0 = 0.f, c1 = 0.f, h0 = 0.f, h1 = 0.f;

        for (int t = 0; t < T_SEQ; ++t) {
            const __bf16* zr = (t & 1) ? z1 : z0;
            __bf16*       zw = (t & 1) ? z0 : z1;
            v8f acc = wmma_matvec_tile(Wtile, KZ, zr, KZ, lane);
            if (extract) {
                // rows j=0..3 -> (f,i,g,o) of unit ua; j=4..7 -> unit ua+1
                float fg0 = sigmoidf_(acc[0] + bb[0]);
                float ig0 = sigmoidf_(acc[1] + bb[1]);
                float gg0 = tanhf    (acc[2] + bb[2]);
                float og0 = sigmoidf_(acc[3] + bb[3]);
                c0 = fg0 * c0 + ig0 * gg0;
                h0 = og0 * tanhf(c0);
                float fg1 = sigmoidf_(acc[4] + bb[4]);
                float ig1 = sigmoidf_(acc[5] + bb[5]);
                float gg1 = tanhf    (acc[6] + bb[6]);
                float og1 = sigmoidf_(acc[7] + bb[7]);
                c1 = fg1 * c1 + ig1 * gg1;
                h1 = og1 * tanhf(c1);
                zw[EMB + ua]     = (__bf16)h0;
                zw[EMB + ua + 1] = (__bf16)h1;
            }
            // one spare lane per wave refreshes one element of x_{t+1}
            if (lane == 2 && t + 1 < T_SEQ)
                zw[wv] = xemb[(size_t)(t + 1) * EMB + wv];
            grid_barrier(bar, NWG);
        }
        if (extract) {
            out[(size_t)T_SEQ * NCH + ua]           = h0;
            out[(size_t)T_SEQ * NCH + ua + 1]       = h1;
            out[(size_t)T_SEQ * NCH + HID + ua]     = c0;
            out[(size_t)T_SEQ * NCH + HID + ua + 1] = c1;
        }
    } else {
        // -------- logits wave (block 64): out[t-1] = Wfc @ h_{t-1} + bfc ----
        const int tile = wv - GTILES;        // 0..7
        const float* bp = bfc_ + tile * 16 + half * 8;
        float bb[8];
#pragma unroll
        for (int j = 0; j < 8; ++j) bb[j] = bp[j];
        const __bf16* Wtile = Wfcb + (size_t)tile * 16 * HID;

        for (int t = 0; t < T_SEQ; ++t) {
            const __bf16* zr = (t & 1) ? z1 : z0;   // h part = h_{t-1}
            v8f acc = wmma_matvec_tile(Wtile, HID, zr + EMB, HID, lane);
            if (extract && t > 0) {
                float* p = out + (size_t)(t - 1) * NCH + tile * 16 + half * 8;
#pragma unroll
                for (int j = 0; j < 8; ++j) p[j] = acc[j] + bb[j];
            }
            grid_barrier(bar, NWG);
        }
        // final logits: h_{T-1} sits in zbuf[T&1] (written at t = T-1)
        const __bf16* zr = (T_SEQ & 1) ? z1 : z0;
        v8f acc = wmma_matvec_tile(Wtile, HID, zr + EMB, HID, lane);
        if (extract) {
            float* p = out + (size_t)(T_SEQ - 1) * NCH + tile * 16 + half * 8;
#pragma unroll
            for (int j = 0; j < 8; ++j) p[j] = acc[j] + bb[j];
        }
    }
}

// ---------------------------------------------------------------------------
extern "C" void kernel_launch(void* const* d_in, const int* in_sizes, int n_in,
                              void* d_out, int out_size, void* d_ws, size_t ws_size,
                              hipStream_t stream) {
    const int*   seq  = (const int*)  d_in[0];
    const float* emb  = (const float*)d_in[1];
    const float* Wf   = (const float*)d_in[2];
    const float* bf_  = (const float*)d_in[3];
    const float* Wi   = (const float*)d_in[4];
    const float* bi_  = (const float*)d_in[5];
    const float* Wg   = (const float*)d_in[6];
    const float* bg_  = (const float*)d_in[7];
    const float* Wo   = (const float*)d_in[8];
    const float* bo_  = (const float*)d_in[9];
    const float* Wfc  = (const float*)d_in[10];
    const float* bfc_ = (const float*)d_in[11];
    float* out = (float*)d_out;

    char* ws = (char*)d_ws;
    size_t off = 0;
    auto carve = [&](size_t bytes) -> char* {
        char* p = ws + off;
        off = (off + bytes + 255) & ~(size_t)255;
        return p;
    };
    __bf16*   Wz   = (__bf16*)  carve((size_t)GROWS * KZ * 2);   // 41.9 MB
    __bf16*   Wfcb = (__bf16*)  carve((size_t)NCH * HID * 2);    //  0.5 MB
    __bf16*   xemb = (__bf16*)  carve((size_t)T_SEQ * EMB * 2);  // 16.8 MB
    __bf16*   z0   = (__bf16*)  carve((size_t)KZ * 2);
    __bf16*   z1   = (__bf16*)  carve((size_t)KZ * 2);
    float*    bz   = (float*)   carve((size_t)GROWS * 4);
    unsigned* bar  = (unsigned*)carve(256);
    (void)ws_size; (void)in_sizes; (void)n_in; (void)out_size;

    k_prep<<<4096, BLK, 0, stream>>>(seq, emb, Wf, Wi, Wg, Wo, Wfc,
                                     bf_, bi_, bg_, bo_,
                                     Wz, Wfcb, xemb, z0, bz, bar);
    k_lstm<<<NWG, BLK, 0, stream>>>(Wz, Wfcb, xemb, z0, z1, bz, bfc_, bar, out);
}
